// SelfAttention_57758720196704
// MI455X (gfx1250) — matrix operand order repaired and checked by reading, not compiled, not run on hardware
//
#include <hip/hip_runtime.h>
#include <hip/hip_bf16.h>

#define BATCH   4
#define SEQ     2048
#define EMBED   1024
#define HEADS   16
#define HDIM    64
#define ROWS    (BATCH * SEQ)       /* 8192 token rows */
#define QK_SCALE 0.125f             /* 64^-0.5 */
#define AQB     64                  /* q rows per attention workgroup */

typedef __bf16 bf16_t;
typedef __attribute__((ext_vector_type(16))) __bf16 v16bf;
typedef __attribute__((ext_vector_type(8)))  __bf16 v8bf;
typedef __attribute__((ext_vector_type(4)))  __bf16 v4bf;
typedef __attribute__((ext_vector_type(8)))  float  v8f;
typedef __attribute__((ext_vector_type(4)))  int    v4i;

union ABfrag { v16bf v; v8bf h[2]; };

// --------------------------------------------------------------------------
// WMMA helpers (CDNA5 wave32 layouts, CDNA5 ISA §7.12.2)
// --------------------------------------------------------------------------
__device__ __forceinline__ v8f wmma_bf16(v16bf a, v16bf b, v8f c) {
  return __builtin_amdgcn_wmma_f32_16x16x32_bf16(
      false, a, false, b, (short)0, c, false, false);
}

// A fragment 16x32 bf16: lane = M, two contiguous 16B loads per lane.
__device__ __forceinline__ v16bf load_afrag(const bf16_t* p, int ld) {
  const int lane = threadIdx.x & 31;
  const bf16_t* r = p + (lane & 15) * ld + (lane >> 4) * 16;
  ABfrag u;
  u.h[0] = *(const v8bf*)(r);
  u.h[1] = *(const v8bf*)(r + 8);
  return u.v;
}

// B fragment 32x16 bf16: lane = N, one contiguous 32B load per lane.
// Works for both global and LDS source pointers.
__device__ __forceinline__ v16bf load_bfrag(const bf16_t* p, int ld) {
  const int lane = threadIdx.x & 31;
  return *(const v16bf*)(p + (lane & 15) * ld + (lane >> 4) * 16);
}

// --------------------------------------------------------------------------
// Async global -> LDS copy (CDNA5 ASYNCcnt path), with sync fallback.
// Probe-confirmed: param 0 is AS1 (global) int4*, param 1 is AS3 (LDS) int4*.
// --------------------------------------------------------------------------
#if __has_builtin(__builtin_amdgcn_global_load_async_to_lds_b128) && \
    __has_builtin(__builtin_amdgcn_s_wait_asynccnt)
#define HAVE_ASYNC_LDS 1
#else
#define HAVE_ASYNC_LDS 0
#endif

typedef __attribute__((address_space(1))) v4i gas_v4i;  // global int4
typedef __attribute__((address_space(3))) v4i las_v4i;  // LDS int4

__device__ __forceinline__ void async_cp16(void* lds, const void* g) {
#if HAVE_ASYNC_LDS
  // Generic LDS pointer's low 32 bits are the LDS byte address (ISA 10.2).
  __builtin_amdgcn_global_load_async_to_lds_b128(
      (gas_v4i*)(uintptr_t)g,
      (las_v4i*)(uintptr_t)(unsigned int)(uintptr_t)lds, 0, 0);
#else
  *(v8bf*)lds = *(const v8bf*)g;
#endif
}

__device__ __forceinline__ void async_wait_all() {
#if HAVE_ASYNC_LDS
  __builtin_amdgcn_s_wait_asynccnt(0);
#endif
}

// --------------------------------------------------------------------------
// Cross-lane xor-max reduction within 16-lane rows via v_permlane16_b32
// (VALU crossbar; keeps the LDS pipe free for matrix fragments).
// --------------------------------------------------------------------------
#if __has_builtin(__builtin_amdgcn_permlane16)
#define HAVE_PERMLANE16 1
#else
#define HAVE_PERMLANE16 0
#endif

__device__ __forceinline__ float permxor16(float v, unsigned lo, unsigned hi, int m) {
#if HAVE_PERMLANE16
  int i = __float_as_int(v);
  i = __builtin_amdgcn_permlane16(i, i, (int)lo, (int)hi, false, false);
  return __int_as_float(i);
#else
  return __shfl_xor(v, m);
#endif
}

__device__ __forceinline__ float redmax16(float v) {
  v = fmaxf(v, permxor16(v, 0x67452301u, 0xEFCDAB89u, 1));
  v = fmaxf(v, permxor16(v, 0x54761032u, 0xDCFEBA98u, 2));
  v = fmaxf(v, permxor16(v, 0x32107654u, 0xBA98FEDCu, 4));
  v = fmaxf(v, permxor16(v, 0xFEDCBA98u, 0x76543210u, 8));
  return v;
}

// --------------------------------------------------------------------------
// f32 -> bf16 cast kernel
// --------------------------------------------------------------------------
__global__ __launch_bounds__(256)
void cvt_f32_bf16(const float* __restrict__ src, bf16_t* __restrict__ dst, int n) {
  int i = (blockIdx.x * 256 + threadIdx.x) * 4;
  if (i + 3 < n) {
    float4 v = *(const float4*)(src + i);
    v4bf o;
    o[0] = (bf16_t)v.x; o[1] = (bf16_t)v.y; o[2] = (bf16_t)v.z; o[3] = (bf16_t)v.w;
    *(v4bf*)(dst + i) = o;
  }
}

// --------------------------------------------------------------------------
// GEMM: Y[M][N] = Xbf16[M][K] @ Wbf16[N][K]^T + bias
// Workgroup = 8 waves -> 128x128 macro-tile; wave = 32x64 (2x4 WMMA accums).
// MODE 0: f32 row-major via non-temporal stores (final output)
// MODE 1: bf16 scattered to [B, H, T, D], scaled by oscale (QKV projections;
//         folds the attention 1/sqrt(d) into the Q projection for free)
// --------------------------------------------------------------------------
template <int MODE>
__global__ __launch_bounds__(256, 2)
void gemm_bf16_wmma(const bf16_t* __restrict__ A, const bf16_t* __restrict__ W,
                    const float* __restrict__ bias, float* __restrict__ outF,
                    bf16_t* __restrict__ outH, int K, int N, float oscale)
{
  const int wave = threadIdx.x >> 5;
  const int row0 = blockIdx.x * 128 + (wave & 3) * 32;
  const int col0 = blockIdx.y * 128 + (wave >> 2) * 64;
  const int lane = threadIdx.x & 31;

  v8f acc[2][4];
  #pragma unroll
  for (int mi = 0; mi < 2; mi++)
    #pragma unroll
    for (int ni = 0; ni < 4; ni++)
      #pragma unroll
      for (int e = 0; e < 8; e++) acc[mi][ni][e] = 0.0f;

  const bf16_t* Ap = A + (size_t)row0 * K;
  const bf16_t* Wp = W + (size_t)col0 * K;

  for (int k0 = 0; k0 < K; k0 += 32) {
    __builtin_prefetch(Ap + k0 + 128, 0, 1);
    __builtin_prefetch(Wp + k0 + 128, 0, 1);
    v16bf af[2], bfr[4];
    af[0] = load_afrag(Ap + k0, K);
    af[1] = load_afrag(Ap + 16 * K + k0, K);
    #pragma unroll
    for (int ni = 0; ni < 4; ni++)
      bfr[ni] = load_bfrag(Wp + ni * 16 * K + k0, K);
    #pragma unroll
    for (int mi = 0; mi < 2; mi++)
      #pragma unroll
      for (int ni = 0; ni < 4; ni++)
        acc[mi][ni] = wmma_bf16(af[mi], bfr[ni], acc[mi][ni]);
  }

  // C layout: VGPR e -> M = e (+8 for high half-wave), N = lane & 15
  const int nloc  = lane & 15;
  const int mbase = (lane >> 4) * 8;
  #pragma unroll
  for (int ni = 0; ni < 4; ni++) {
    const int col = col0 + ni * 16 + nloc;
    const float bv = bias[col];
    #pragma unroll
    for (int mi = 0; mi < 2; mi++) {
      #pragma unroll
      for (int e = 0; e < 8; e++) {
        const int row = row0 + mi * 16 + mbase + e;
        const float val = acc[mi][ni][e] + bv;
        if (MODE == 0) {
          __builtin_nontemporal_store(val, &outF[(size_t)row * N + col]);
        } else {
          const int b = row >> 11, t = row & (SEQ - 1);
          const int h = col >> 6,  d = col & (HDIM - 1);
          outH[((((size_t)b * HEADS + h) * SEQ + t) << 6) + d] = (bf16_t)(val * oscale);
        }
      }
    }
  }
}

// --------------------------------------------------------------------------
// Flash attention core.
// Workgroup = 4 waves sharing one (b,h); wave w owns q rows [t0, t0+16).
// K and V tiles (4 KB each) are double-buffered in LDS and staged with
// GLOBAL_LOAD_ASYNC_TO_LDS_B128 one tile ahead: the DMA for tile i+1 is
// issued right after tile i's Q.K^T and overlaps softmax + P.V.  One
// workgroup barrier per iteration.  Row sums computed by an extra WMMA
// against a ones fragment (XDL pipe) instead of permlane butterflies.
// --------------------------------------------------------------------------
__global__ __launch_bounds__(128)
void attn_flash_wmma(const bf16_t* __restrict__ Q, const bf16_t* __restrict__ Kt,
                     const bf16_t* __restrict__ V, bf16_t* __restrict__ Ctx)
{
  const int bh   = blockIdx.x / (SEQ / AQB);
  const int qb   = blockIdx.x % (SEQ / AQB);
  const int wave = threadIdx.x >> 5;
  const int lane = threadIdx.x & 31;
  const int t0   = qb * AQB + wave * 16;

  const bf16_t* Qp = Q + ((size_t)bh * SEQ + t0) * HDIM;
  const bf16_t* Kp = Kt + (size_t)bh * SEQ * HDIM;
  const bf16_t* Vp = V + (size_t)bh * SEQ * HDIM;

  __shared__ __align__(128) bf16_t Klds[2][32 * HDIM];   // K tile, row-major [s][d]
  __shared__ __align__(128) bf16_t Vlds[2][32 * HDIM];   // V tile, row-major [s][d]
  __shared__ __align__(128) bf16_t Pbuf[4][16 * 32];     // per-wave probs [m][s]

  const int nloc  = lane & 15;
  const int half  = lane >> 4;
  const int mbase = half * 8;
  bf16_t* pb = &Pbuf[wave][0];

  const v16bf aq0 = load_afrag(Qp, HDIM);        // d = 0..31 (already *QK_SCALE)
  const v16bf aq1 = load_afrag(Qp + 32, HDIM);   // d = 32..63

  v16bf ones;
  #pragma unroll
  for (int k = 0; k < 16; k++) ones[k] = (bf16_t)1.0f;

  float rm[8];
  v8f acc[4], accS;
  #pragma unroll
  for (int e = 0; e < 8; e++) { rm[e] = -1e30f; accS[e] = 0.0f; }
  #pragma unroll
  for (int ni = 0; ni < 4; ni++)
    #pragma unroll
    for (int e = 0; e < 8; e++) acc[ni][e] = 0.0f;

  // Stage K+V tile for s0 into buffer b2 (8 KB = 4 x (128 threads * 16B)).
  auto stage = [&](int b2, int s0) {
    const char* gk = (const char*)(Kp + (size_t)s0 * HDIM) + threadIdx.x * 16;
    const char* gv = (const char*)(Vp + (size_t)s0 * HDIM) + threadIdx.x * 16;
    char* lk = (char*)&Klds[b2][0] + threadIdx.x * 16;
    char* lv = (char*)&Vlds[b2][0] + threadIdx.x * 16;
    async_cp16(lk, gk);
    async_cp16(lk + 2048, gk + 2048);
    async_cp16(lv, gv);
    async_cp16(lv + 2048, gv + 2048);
  };

  stage(0, 0);

  int buf = 0;
  for (int s0 = 0; s0 < SEQ; s0 += 32) {
    // Tile resident (own async done) + visible (all waves arrived).
    async_wait_all();
    __syncthreads();

    // ---- scores: 16x32 tile, two 16x16 WMMA accums over K-dim = HDIM ----
    v8f sc[2];
    #pragma unroll
    for (int j = 0; j < 2; j++) {
      const bf16_t* kb = &Klds[buf][j * 16 * HDIM];
      v8f z;
      #pragma unroll
      for (int e = 0; e < 8; e++) z[e] = 0.0f;
      sc[j] = wmma_bf16(aq0, load_bfrag(kb, HDIM), z);
      sc[j] = wmma_bf16(aq1, load_bfrag(kb + 32, HDIM), sc[j]);
    }

    // Kick off DMA for the next tile; overlaps softmax + P.V below.
    if (s0 + 32 < SEQ) stage(buf ^ 1, s0 + 32);

    // ---- online softmax (row stats shared across the 16-lane N group) ----
    #pragma unroll
    for (int e = 0; e < 8; e++) {
      const float a0 = sc[0][e];
      const float a1 = sc[1][e];
      const float tmax = redmax16(fmaxf(a0, a1));
      const float mnew = fmaxf(rm[e], tmax);
      const float f = __expf(rm[e] - mnew);
      rm[e] = mnew;
      const float p0 = __expf(a0 - mnew);
      const float p1 = __expf(a1 - mnew);
      accS[e] *= f;
      #pragma unroll
      for (int ni = 0; ni < 4; ni++) acc[ni][e] *= f;
      const int mrow = mbase + e;
      pb[mrow * 32 + nloc]      = (bf16_t)p0;
      pb[mrow * 32 + 16 + nloc] = (bf16_t)p1;
    }

    // ---- P (A-layout, per-wave LDS); row sums via ones-fragment WMMA ----
    ABfrag pf;
    pf.h[0] = *(const v8bf*)(&pb[nloc * 32 + half * 16]);
    pf.h[1] = *(const v8bf*)(&pb[nloc * 32 + half * 16 + 8]);
    accS = wmma_bf16(pf.v, ones, accS);

    // ---- P x V (B-layout gathered from row-major LDS) ----
    const bf16_t* vt = &Vlds[buf][0];
    #pragma unroll
    for (int ni = 0; ni < 4; ni++) {
      const int d = ni * 16 + nloc;
      v16bf bv;
      #pragma unroll
      for (int k = 0; k < 16; k++)
        bv[k] = vt[(half * 16 + k) * HDIM + d];
      acc[ni] = wmma_bf16(pf.v, bv, acc[ni]);
    }
    buf ^= 1;
  }

  // ---- normalize and write ctx as [B, T, E] bf16 ----
  const int b = bh >> 4, h = bh & (HEADS - 1);
  #pragma unroll
  for (int e = 0; e < 8; e++) {
    const float inv = 1.0f / accS[e];
    const int t = t0 + mbase + e;
    bf16_t* orow = Ctx + ((size_t)b * SEQ + t) * EMBED + h * HDIM;
    #pragma unroll
    for (int ni = 0; ni < 4; ni++)
      orow[ni * 16 + nloc] = (bf16_t)(acc[ni][e] * inv);
  }
}

// --------------------------------------------------------------------------
// Launcher
// --------------------------------------------------------------------------
extern "C" void kernel_launch(void* const* d_in, const int* in_sizes, int n_in,
                              void* d_out, int out_size, void* d_ws, size_t ws_size,
                              hipStream_t stream) {
  (void)in_sizes; (void)n_in; (void)out_size; (void)ws_size;
  const float* X  = (const float*)d_in[0];
  const float* Wq = (const float*)d_in[1];
  const float* bq = (const float*)d_in[2];
  const float* Wk = (const float*)d_in[3];
  const float* bk = (const float*)d_in[4];
  const float* Wv = (const float*)d_in[5];
  const float* bv = (const float*)d_in[6];
  const float* Wo = (const float*)d_in[7];
  const float* bo = (const float*)d_in[8];
  float* out = (float*)d_out;

  char* ws = (char*)d_ws;
  size_t off = 0;
  auto carve = [&](size_t bytes) -> char* {
    char* p = ws + off;
    off += (bytes + 255) & ~(size_t)255;
    return p;
  };
  const size_t actB = (size_t)ROWS * EMBED * sizeof(bf16_t);   // 16 MB
  const size_t wB   = (size_t)EMBED * EMBED * sizeof(bf16_t);  // 2 MB
  bf16_t* Xb  = (bf16_t*)carve(actB);
  bf16_t* Wqb = (bf16_t*)carve(wB);
  bf16_t* Wkb = (bf16_t*)carve(wB);
  bf16_t* Wvb = (bf16_t*)carve(wB);
  bf16_t* Wob = (bf16_t*)carve(wB);
  bf16_t* Qb  = (bf16_t*)carve(actB);
  bf16_t* Kb  = (bf16_t*)carve(actB);
  bf16_t* Vb  = (bf16_t*)carve(actB);
  bf16_t* Cb  = (bf16_t*)carve(actB);

  const int nAct = ROWS * EMBED;     // 8M elems
  const int nW   = EMBED * EMBED;    // 1M elems
  cvt_f32_bf16<<<nAct / (256 * 4), 256, 0, stream>>>(X,  Xb,  nAct);
  cvt_f32_bf16<<<nW   / (256 * 4), 256, 0, stream>>>(Wq, Wqb, nW);
  cvt_f32_bf16<<<nW   / (256 * 4), 256, 0, stream>>>(Wk, Wkb, nW);
  cvt_f32_bf16<<<nW   / (256 * 4), 256, 0, stream>>>(Wv, Wvb, nW);
  cvt_f32_bf16<<<nW   / (256 * 4), 256, 0, stream>>>(Wo, Wob, nW);

  dim3 gg(ROWS / 128, EMBED / 128);
  gemm_bf16_wmma<1><<<gg, 256, 0, stream>>>(Xb, Wqb, bq, nullptr, Qb, EMBED, EMBED, QK_SCALE);
  gemm_bf16_wmma<1><<<gg, 256, 0, stream>>>(Xb, Wkb, bk, nullptr, Kb, EMBED, EMBED, 1.0f);
  gemm_bf16_wmma<1><<<gg, 256, 0, stream>>>(Xb, Wvb, bv, nullptr, Vb, EMBED, EMBED, 1.0f);

  attn_flash_wmma<<<BATCH * HEADS * (SEQ / AQB), 128, 0, stream>>>(Qb, Kb, Vb, Cb);

  gemm_bf16_wmma<0><<<gg, 256, 0, stream>>>(Cb, Wob, bo, out, nullptr, EMBED, EMBED, 1.0f);
}
